// HEAGNN_41729902248046
// MI455X (gfx1250) — compile-verified, matrix operand order checked by reading, not run on hardware
//
#include <hip/hip_runtime.h>
#include <hip/hip_bf16.h>
#include <math.h>
#include <stdint.h>

// ---- problem constants ----
constexpr int kH  = 128;
constexpr int kL  = 4;
constexpr int kNH = 4;
constexpr int kNN = 16384;
constexpr int kE  = 262144;
constexpr int kB  = 32;
constexpr int kN  = 512;

typedef __bf16 bf16_t;
typedef __attribute__((ext_vector_type(16))) __bf16 v16bf;
typedef __attribute__((ext_vector_type(8)))  __bf16 v8bf;
typedef __attribute__((ext_vector_type(8)))  float  v8f;

__device__ inline v8f vzero8() {
    v8f z;
#pragma unroll
    for (int i = 0; i < 8; ++i) z[i] = 0.f;
    return z;
}

// Build a 16-element bf16 WMMA fragment from two aligned 8-element chunks.
// CDNA5 16-bit A layout: lane (l<16) holds K = {khalf..khalf+7, 16+khalf..16+khalf+7}
// with khalf = (l>=16)*8; B is loaded with the mirrored (column-indexed) pattern.
__device__ inline v16bf load_frag(const bf16_t* p0, const bf16_t* p1) {
    union { v16bf v; v8bf h[2]; } u;
    u.h[0] = *(const v8bf*)p0;
    u.h[1] = *(const v8bf*)p1;
    return u.v;
}

// =====================================================================
// Row embedding: out = relu(LN(x[r,:K] @ W[K,128] + b)) ; one block per row
// =====================================================================
__global__ __launch_bounds__(128) void k_embed(const float* __restrict__ X, int K,
                                               const float* __restrict__ W,
                                               const float* __restrict__ b,
                                               const float* __restrict__ g,
                                               const float* __restrict__ beta,
                                               float* __restrict__ outF,
                                               bf16_t* __restrict__ outB) {
    const long long r = blockIdx.x;
    const int c = threadIdx.x;
    const float* x = X + r * K;
    float u = b[c];
    for (int k = 0; k < K; ++k) u += x[k] * W[k * kH + c];
    __shared__ float red[128];
    red[c] = u; __syncthreads();
    for (int st = 64; st > 0; st >>= 1) { if (c < st) red[c] += red[c + st]; __syncthreads(); }
    const float mean = red[0] / 128.f; __syncthreads();
    const float dv = u - mean;
    red[c] = dv * dv; __syncthreads();
    for (int st = 64; st > 0; st >>= 1) { if (c < st) red[c] += red[c + st]; __syncthreads(); }
    const float var = red[0] / 128.f;
    float y = dv * rsqrtf(var + 1e-5f) * g[c] + beta[c];
    y = fmaxf(y, 0.f);
    const long long o = r * kH + c;
    if (outF) outF[o] = y;
    if (outB) outB[o] = (bf16_t)y;
}

// =====================================================================
// Row LayerNorm only (f32 in -> bf16 out); one block per row
// =====================================================================
__global__ __launch_bounds__(128) void k_lnrow(const float* __restrict__ X,
                                               const float* __restrict__ g,
                                               const float* __restrict__ beta,
                                               bf16_t* __restrict__ outB) {
    const long long r = blockIdx.x;
    const int c = threadIdx.x;
    const float u = X[r * kH + c];
    __shared__ float red[128];
    red[c] = u; __syncthreads();
    for (int st = 64; st > 0; st >>= 1) { if (c < st) red[c] += red[c + st]; __syncthreads(); }
    const float mean = red[0] / 128.f; __syncthreads();
    const float dv = u - mean;
    red[c] = dv * dv; __syncthreads();
    for (int st = 64; st > 0; st >>= 1) { if (c < st) red[c] += red[c + st]; __syncthreads(); }
    const float var = red[0] / 128.f;
    outB[r * kH + c] = (bf16_t)(dv * rsqrtf(var + 1e-5f) * g[c] + beta[c]);
}

// =====================================================================
// Generic WMMA GEMM: C[M,128] = A_bf16[M,128] @ W_f32[128,128]
// epilogue: (+bias)*oscale (+resid) | or LN(acc+bias)*g+beta ; optional relu
// writes f32 and/or bf16. Block = 64 rows x 128 cols, 256 threads (8 waves).
// =====================================================================
__global__ __launch_bounds__(256) void k_gemm128(const bf16_t* __restrict__ A,
                                                 const float* __restrict__ W,
                                                 const float* __restrict__ bias,
                                                 const float* __restrict__ resid,
                                                 const float* __restrict__ g,
                                                 const float* __restrict__ beta,
                                                 float* __restrict__ outF,
                                                 bf16_t* __restrict__ outB,
                                                 float oscale, int relu) {
    __shared__ __align__(16) bf16_t Wt[128 * 136];   // W transposed (bf16), 34816 B
    __shared__ float rowm[64], rowr[64];
    float* Cs = (float*)Wt;                          // LN staging aliases Wt (64*132*4 B)

    const int tid  = threadIdx.x;
    const int lane = tid & 31;
    const int wv   = tid >> 5;
    const int wm   = wv >> 1;      // 0..3 : 16-row tile
    const int wn   = wv & 1;       // 0..1 : 64-col half
    const long long rowBlock = (long long)blockIdx.x * 64;

    for (int idx = tid; idx < 128 * 128; idx += 256) {
        const int k = idx >> 7, n = idx & 127;
        Wt[n * 136 + k] = (bf16_t)W[idx];
    }
    __syncthreads();

    v8f acc[4];
#pragma unroll
    for (int nt = 0; nt < 4; ++nt) acc[nt] = vzero8();

    const int mrow  = wm * 16 + (lane & 15);
    const int khalf = (lane >> 4) * 8;
    const bf16_t* Arow = A + (rowBlock + mrow) * kH;

#pragma unroll
    for (int kk = 0; kk < 128; kk += 32) {
        const v16bf a = load_frag(Arow + kk + khalf, Arow + kk + 16 + khalf);
        // Stage all 4 B fragments first (distinct registers) so the LDS loads
        // clause together and the 4 independent WMMAs issue back-to-back.
        v16bf bfr[4];
#pragma unroll
        for (int nt = 0; nt < 4; ++nt) {
            const int col = wn * 64 + nt * 16 + (lane & 15);
            bfr[nt] = load_frag(&Wt[col * 136 + kk + khalf],
                                &Wt[col * 136 + kk + 16 + khalf]);
        }
#pragma unroll
        for (int nt = 0; nt < 4; ++nt) {
            acc[nt] = __builtin_amdgcn_wmma_f32_16x16x32_bf16(
                false, a, false, bfr[nt], (short)0, acc[nt], false, false);
        }
    }

    if (g == nullptr) {
        // direct epilogue from accumulators
#pragma unroll
        for (int nt = 0; nt < 4; ++nt) {
            const int col = wn * 64 + nt * 16 + (lane & 15);
            const float bv = bias ? bias[col] : 0.f;
#pragma unroll
            for (int r = 0; r < 8; ++r) {
                const long long row = rowBlock + wm * 16 + r + ((lane >> 4) * 8);
                const long long o = row * kH + col;
                float v = (acc[nt][r] + bv) * oscale;
                if (resid) v += resid[o];
                if (relu)  v = fmaxf(v, 0.f);
                if (outF) outF[o] = v;
                if (outB) outB[o] = (bf16_t)v;
            }
        }
    } else {
        // fused LayerNorm epilogue: stage tile in LDS (reuses Wt space)
        __syncthreads();
#pragma unroll
        for (int nt = 0; nt < 4; ++nt) {
            const int col = wn * 64 + nt * 16 + (lane & 15);
            const float bv = bias ? bias[col] : 0.f;
#pragma unroll
            for (int r = 0; r < 8; ++r) {
                const int rr = wm * 16 + r + ((lane >> 4) * 8);
                Cs[rr * 132 + col] = acc[nt][r] + bv;
            }
        }
        __syncthreads();
        if (tid < 64) {
            float s = 0.f;
            for (int c = 0; c < 128; ++c) s += Cs[tid * 132 + c];
            const float mean = s / 128.f;
            float vv = 0.f;
            for (int c = 0; c < 128; ++c) { const float d = Cs[tid * 132 + c] - mean; vv += d * d; }
            rowm[tid] = mean;
            rowr[tid] = rsqrtf(vv / 128.f + 1e-5f);
        }
        __syncthreads();
        for (int idx = tid; idx < 64 * 128; idx += 256) {
            const int r = idx >> 7, c = idx & 127;
            float v = (Cs[r * 132 + c] - rowm[r]) * rowr[r] * g[c] + beta[c];
            if (relu) v = fmaxf(v, 0.f);
            const long long o = (rowBlock + r) * kH + c;
            if (outF) outF[o] = v;
            if (outB) outB[o] = (bf16_t)v;
        }
    }
}

// =====================================================================
// Edge combine: gate = sigmoid(P1[e]+P2[dst]+P3[src]); msg = gate*nmlp[src]
// scatter-add into agg[dst] with f32 global atomics. One block per edge.
// =====================================================================
__global__ __launch_bounds__(128) void k_edge(const int* __restrict__ eidx,
                                              const float* __restrict__ P1,
                                              const float* __restrict__ P2,
                                              const float* __restrict__ P3,
                                              const float* __restrict__ nmlp,
                                              float* __restrict__ agg) {
    const long long e = blockIdx.x;
    const int c = threadIdx.x;
    const long long s = eidx[e];
    const long long d = eidx[(long long)kE + e];
    const float logit = P1[e * kH + c] + P2[d * kH + c] + P3[s * kH + c];
    const float gate = 1.f / (1.f + __expf(-logit));
    const float msg = gate * nmlp[s * kH + c];
    __hip_atomic_fetch_add(&agg[d * kH + c], msg, __ATOMIC_RELAXED, __HIP_MEMORY_SCOPE_AGENT);
}

// h += agg + nmlp (residual conv update)
__global__ __launch_bounds__(256) void k_add3(float* __restrict__ h,
                                              const float* __restrict__ agg,
                                              const float* __restrict__ nmlp) {
    const long long i = (long long)blockIdx.x * 256 + threadIdx.x;
    h[i] += agg[i] + nmlp[i];
}

// =====================================================================
// Attention scores: S[b,hd,i,j] = Q_row(i) . K_row(j)  (scale folded into Q)
// hd dim = 32 -> exactly one 16x16x32 WMMA per 16x16 tile.
// grid = B*NH*8*8 blocks, each = 64x64 tile of S for one (b,head).
// =====================================================================
__global__ __launch_bounds__(256) void k_scores(const bf16_t* __restrict__ Qb,
                                                const bf16_t* __restrict__ Kb,
                                                float* __restrict__ S) {
    const int bi = blockIdx.x;
    const int jt = bi & 7, it = (bi >> 3) & 7, head = (bi >> 6) & 3, b = bi >> 8;
    const int tid = threadIdx.x, lane = tid & 31;
    const int wv = tid >> 5, wm = wv >> 1, wn = wv & 1;
    const int khalf = (lane >> 4) * 8;
    const int i0 = it * 64 + wm * 16;
    const int j0 = jt * 64 + wn * 32;

    const bf16_t* qrow = Qb + ((long long)(b * kN + i0 + (lane & 15))) * kH + head * 32;
    const v16bf a = load_frag(qrow + khalf, qrow + 16 + khalf);

    v16bf bfr[2];
#pragma unroll
    for (int nt = 0; nt < 2; ++nt) {
        const bf16_t* krow = Kb + ((long long)(b * kN + j0 + nt * 16 + (lane & 15))) * kH + head * 32;
        bfr[nt] = load_frag(krow + khalf, krow + 16 + khalf);
    }
    v8f acc[2]; acc[0] = vzero8(); acc[1] = vzero8();
#pragma unroll
    for (int nt = 0; nt < 2; ++nt) {
        acc[nt] = __builtin_amdgcn_wmma_f32_16x16x32_bf16(
            false, a, false, bfr[nt], (short)0, acc[nt], false, false);
    }
    const long long base = ((long long)(b * kNH + head)) * kN * kN;
#pragma unroll
    for (int nt = 0; nt < 2; ++nt) {
        const int j = j0 + nt * 16 + (lane & 15);
#pragma unroll
        for (int r = 0; r < 8; ++r) {
            const int i = i0 + r + ((lane >> 4) * 8);
            S[base + (long long)i * kN + j] = acc[nt][r];
        }
    }
}

// Row softmax over 512 (f32 in -> bf16 out). One block per row, 256 threads.
__global__ __launch_bounds__(256) void k_softmax(const float* __restrict__ S,
                                                 bf16_t* __restrict__ P) {
    const long long row = blockIdx.x;
    const float* s = S + row * kN;
    bf16_t* p = P + row * kN;
    const int tid = threadIdx.x;
    const float v0 = s[tid], v1 = s[tid + 256];
    __shared__ float red[256];
    red[tid] = fmaxf(v0, v1); __syncthreads();
    for (int st = 128; st > 0; st >>= 1) { if (tid < st) red[tid] = fmaxf(red[tid], red[tid + st]); __syncthreads(); }
    const float m = red[0]; __syncthreads();
    const float e0 = __expf(v0 - m), e1 = __expf(v1 - m);
    red[tid] = e0 + e1; __syncthreads();
    for (int st = 128; st > 0; st >>= 1) { if (tid < st) red[tid] += red[tid + st]; __syncthreads(); }
    const float inv = 1.f / red[0];
    p[tid] = (bf16_t)(e0 * inv);
    p[tid + 256] = (bf16_t)(e1 * inv);
}

// =====================================================================
// O = P @ V per (b,head): A = P[512,512] bf16, V staged transposed in LDS.
// grid = B*NH*8 blocks, each 64 i-rows x 32 d-cols; 16 WMMA k-steps split
// into two independent accumulator chains for better pipelining.
// =====================================================================
__global__ __launch_bounds__(256) void k_pv(const bf16_t* __restrict__ P,
                                            const bf16_t* __restrict__ Vb,
                                            bf16_t* __restrict__ O) {
    const int bi = blockIdx.x;
    const int it = bi & 7, head = (bi >> 3) & 3, b = bi >> 5;
    __shared__ __align__(16) bf16_t Vt[32 * 520];
    const int tid = threadIdx.x;
    for (int idx = tid; idx < 32 * 512; idx += 256) {
        const int d = idx & 31, j = idx >> 5;
        Vt[d * 520 + j] = Vb[((long long)(b * kN + j)) * kH + head * 32 + d];
    }
    __syncthreads();

    const int lane = tid & 31, wv = tid >> 5, wm = wv >> 1, wn = wv & 1;
    const int khalf = (lane >> 4) * 8;
    const int i0 = it * 64 + wm * 16;
    const int dcol = wn * 16 + (lane & 15);
    const bf16_t* prow = P + (((long long)(b * kNH + head)) * kN + i0 + (lane & 15)) * kN;

    v8f acc0 = vzero8(), acc1 = vzero8();
#pragma unroll
    for (int kk = 0; kk < 512; kk += 64) {
        const v16bf a0 = load_frag(prow + kk + khalf, prow + kk + 16 + khalf);
        const v16bf b0 = load_frag(&Vt[dcol * 520 + kk + khalf],
                                   &Vt[dcol * 520 + kk + 16 + khalf]);
        const v16bf a1 = load_frag(prow + kk + 32 + khalf, prow + kk + 48 + khalf);
        const v16bf b1 = load_frag(&Vt[dcol * 520 + kk + 32 + khalf],
                                   &Vt[dcol * 520 + kk + 48 + khalf]);
        acc0 = __builtin_amdgcn_wmma_f32_16x16x32_bf16(
            false, a0, false, b0, (short)0, acc0, false, false);
        acc1 = __builtin_amdgcn_wmma_f32_16x16x32_bf16(
            false, a1, false, b1, (short)0, acc1, false, false);
    }
#pragma unroll
    for (int r = 0; r < 8; ++r) {
        const int i = i0 + r + ((lane >> 4) * 8);
        O[((long long)(b * kN + i)) * kH + head * 32 + dcol] = (bf16_t)(acc0[r] + acc1[r]);
    }
}

// Per-graph mean/max pooling (equal-size, sorted graphs)
__global__ __launch_bounds__(128) void k_pool(const float* __restrict__ h,
                                              float* __restrict__ xmean,
                                              float* __restrict__ xmax) {
    const int b = blockIdx.x, c = threadIdx.x;
    float s = 0.f, mx = -3.4e38f;
    for (int r = 0; r < kN; ++r) {
        const float v = h[((long long)(b * kN + r)) * kH + c];
        s += v; mx = fmaxf(mx, v);
    }
    xmean[b * kH + c] = s / (float)kN;
    xmax[b * kH + c] = mx;
}

// ---------------- tiny readout MLP (B=32) ----------------
__device__ inline float blk_sum(float v, int tid, float* red) {
    red[tid] = v; __syncthreads();
    for (int st = 128; st > 0; st >>= 1) { if (tid < st) red[tid] += red[tid + st]; __syncthreads(); }
    const float r = red[0]; __syncthreads();
    return r;
}

__global__ __launch_bounds__(256) void k_readout(
    const float* __restrict__ xmean, const float* __restrict__ xmax, const float* __restrict__ af,
    const float* amw1, const float* amb1, const float* amg, const float* ambt,
    const float* amw2, const float* amb2,
    const float* ow1, const float* ob1, const float* og1, const float* obt1,
    const float* ow2, const float* ob2, const float* og2, const float* obt2,
    const float* ow3, const float* ob3, const float* ow4, const float* ob4,
    float* __restrict__ out) {
    __shared__ float comb[512], t1[256], t2[128], t3[64], ar[128], red[256];
    const int tid = threadIdx.x;
    for (int b = 0; b < kB; ++b) {
        // additional-features: relu(LN(af@am_w1+b))@am_w2+b2
        float u = 0.f;
        if (tid < 128) {
            u = amb1[tid];
            for (int k = 0; k < 8; ++k) u += af[b * 8 + k] * amw1[k * 128 + tid];
        }
        float mean = blk_sum(tid < 128 ? u : 0.f, tid, red) / 128.f;
        float dv = (tid < 128) ? (u - mean) : 0.f;
        float var = blk_sum(dv * dv, tid, red) / 128.f;
        if (tid < 128) ar[tid] = fmaxf(dv * rsqrtf(var + 1e-5f) * amg[tid] + ambt[tid], 0.f);
        __syncthreads();
        if (tid < 128) {
            float a2 = amb2[tid];
            for (int k = 0; k < 128; ++k) a2 += ar[k] * amw2[k * 128 + tid];
            comb[tid]       = xmean[b * 128 + tid];
            comb[128 + tid] = xmax[b * 128 + tid];
            comb[256 + tid] = xmean[b * 128 + tid];
            comb[384 + tid] = a2;
        }
        __syncthreads();
        // t1 = relu(LN(comb@o_w1+b1))   [256]
        float u1 = ob1[tid];
        for (int k = 0; k < 512; ++k) u1 += comb[k] * ow1[k * 256 + tid];
        mean = blk_sum(u1, tid, red) / 256.f;
        dv = u1 - mean;
        var = blk_sum(dv * dv, tid, red) / 256.f;
        t1[tid] = fmaxf(dv * rsqrtf(var + 1e-5f) * og1[tid] + obt1[tid], 0.f);
        __syncthreads();
        // t2 = relu(LN(t1@o_w2+b2))    [128]
        float u2 = 0.f;
        if (tid < 128) {
            u2 = ob2[tid];
            for (int k = 0; k < 256; ++k) u2 += t1[k] * ow2[k * 128 + tid];
        }
        mean = blk_sum(tid < 128 ? u2 : 0.f, tid, red) / 128.f;
        dv = (tid < 128) ? (u2 - mean) : 0.f;
        var = blk_sum(dv * dv, tid, red) / 128.f;
        if (tid < 128) t2[tid] = fmaxf(dv * rsqrtf(var + 1e-5f) * og2[tid] + obt2[tid], 0.f);
        __syncthreads();
        // t3 = relu(t2@o_w3+b3)        [64]
        if (tid < 64) {
            float u3 = ob3[tid];
            for (int k = 0; k < 128; ++k) u3 += t2[k] * ow3[k * 64 + tid];
            t3[tid] = fmaxf(u3, 0.f);
        }
        __syncthreads();
        const float u4 = (tid < 64) ? t3[tid] * ow4[tid] : 0.f;
        const float s = blk_sum(u4, tid, red);
        if (tid == 0) out[b] = s + ob4[0];
        __syncthreads();
    }
}

// =====================================================================
extern "C" void kernel_launch(void* const* d_in, const int* in_sizes, int n_in,
                              void* d_out, int out_size, void* d_ws, size_t ws_size,
                              hipStream_t stream) {
    (void)in_sizes; (void)n_in; (void)out_size; (void)ws_size;
    const float* x       = (const float*)d_in[0];
    const float* eattr   = (const float*)d_in[1];
    const float* af      = (const float*)d_in[2];
    const float* ne_w    = (const float*)d_in[3];
    const float* ne_b    = (const float*)d_in[4];
    const float* ne_g    = (const float*)d_in[5];
    const float* ne_bt   = (const float*)d_in[6];
    const float* ee_w    = (const float*)d_in[7];
    const float* ee_b    = (const float*)d_in[8];
    const float* ee_g    = (const float*)d_in[9];
    const float* ee_bt   = (const float*)d_in[10];
    const float* nm_w1   = (const float*)d_in[11];
    const float* nm_b1   = (const float*)d_in[12];
    const float* nm_g    = (const float*)d_in[13];
    const float* nm_bt   = (const float*)d_in[14];
    const float* nm_w2   = (const float*)d_in[15];
    const float* nm_b2   = (const float*)d_in[16];
    const float* gw      = (const float*)d_in[17];
    const float* gb      = (const float*)d_in[18];
    const float* at_g    = (const float*)d_in[19];
    const float* at_bt   = (const float*)d_in[20];
    const float* qw      = (const float*)d_in[21];
    const float* kw      = (const float*)d_in[22];
    const float* vw      = (const float*)d_in[23];
    const float* ow      = (const float*)d_in[24];
    const float* qb      = (const float*)d_in[25];
    const float* kb      = (const float*)d_in[26];
    const float* vb      = (const float*)d_in[27];
    const float* ob      = (const float*)d_in[28];
    const float* am_w1   = (const float*)d_in[29];
    const float* am_b1   = (const float*)d_in[30];
    const float* am_g    = (const float*)d_in[31];
    const float* am_bt   = (const float*)d_in[32];
    const float* am_w2   = (const float*)d_in[33];
    const float* am_b2   = (const float*)d_in[34];
    const float* o_w1    = (const float*)d_in[35];
    const float* o_b1    = (const float*)d_in[36];
    const float* o_g1    = (const float*)d_in[37];
    const float* o_bt1   = (const float*)d_in[38];
    const float* o_w2    = (const float*)d_in[39];
    const float* o_b2    = (const float*)d_in[40];
    const float* o_g2    = (const float*)d_in[41];
    const float* o_bt2   = (const float*)d_in[42];
    const float* o_w3    = (const float*)d_in[43];
    const float* o_b3    = (const float*)d_in[44];
    const float* o_w4    = (const float*)d_in[45];
    const float* o_b4    = (const float*)d_in[46];
    const int*   eidx    = (const int*)d_in[47];

    // ---- workspace carve (all 256B aligned) ----
    uintptr_t p = (uintptr_t)d_ws;
    auto take = [&](size_t bytes) -> void* {
        uintptr_t q = (p + 255) & ~(uintptr_t)255;
        p = q + bytes;
        return (void*)q;
    };
    float*  h    = (float*)take((size_t)kNN * kH * 4);
    bf16_t* hb   = (bf16_t*)take((size_t)kNN * kH * 2);
    bf16_t* eab  = (bf16_t*)take((size_t)kE * kH * 2);
    bf16_t* tb   = (bf16_t*)take((size_t)kNN * kH * 2);
    bf16_t* hnb  = (bf16_t*)take((size_t)kNN * kH * 2);
    float*  nmlp = (float*)take((size_t)kNN * kH * 4);
    float*  P2   = (float*)take((size_t)kNN * kH * 4);
    float*  P3   = (float*)take((size_t)kNN * kH * 4);
    float*  agg  = (float*)take((size_t)kNN * kH * 4);
    float*  BIG1 = (float*)take((size_t)kE * kH * 4);   // P1, then attention scores S
    bf16_t* BIG2 = (bf16_t*)take((size_t)kE * kH * 2);  // softmax probabilities (bf16)
    bf16_t* Qb   = (bf16_t*)take((size_t)kNN * kH * 2);
    bf16_t* Kb   = (bf16_t*)take((size_t)kNN * kH * 2);
    bf16_t* Vb   = (bf16_t*)take((size_t)kNN * kH * 2);
    bf16_t* Ob   = (bf16_t*)take((size_t)kNN * kH * 2);
    float*  xmean = (float*)take((size_t)kB * kH * 4);
    float*  xmax  = (float*)take((size_t)kB * kH * 4);

    const float qscale = 0.17677669529663687f; // 1/sqrt(hd=32)

    // embeddings
    k_embed<<<kNN, 128, 0, stream>>>(x, 5, ne_w, ne_b, ne_g, ne_bt, h, hb);
    k_embed<<<kE, 128, 0, stream>>>(eattr, 3, ee_w, ee_b, ee_g, ee_bt, nullptr, eab);

    for (int i = 0; i < kL; ++i) {
        const size_t mm = (size_t)i * kH * kH;
        const size_t mv = (size_t)i * kH;
        // node_mlp(h): t = relu(LN(h@W1+b1)) ; nmlp = t@W2+b2
        k_gemm128<<<kNN / 64, 256, 0, stream>>>(hb, nm_w1 + mm, nm_b1 + mv, nullptr,
                                                nm_g + mv, nm_bt + mv, nullptr, tb, 1.f, 1);
        k_gemm128<<<kNN / 64, 256, 0, stream>>>(tb, nm_w2 + mm, nm_b2 + mv, nullptr,
                                                nullptr, nullptr, nmlp, nullptr, 1.f, 0);
        // gate decomposition: P1 = ea@gw[0:H]+gb ; P2 = h@gw[H:2H] ; P3 = h@gw[2H:3H]
        const float* gwi = gw + (size_t)i * 3 * kH * kH;
        k_gemm128<<<kE / 64, 256, 0, stream>>>(eab, gwi, gb + mv, nullptr,
                                               nullptr, nullptr, BIG1, nullptr, 1.f, 0);
        k_gemm128<<<kNN / 64, 256, 0, stream>>>(hb, gwi + (size_t)kH * kH, nullptr, nullptr,
                                                nullptr, nullptr, P2, nullptr, 1.f, 0);
        k_gemm128<<<kNN / 64, 256, 0, stream>>>(hb, gwi + (size_t)2 * kH * kH, nullptr, nullptr,
                                                nullptr, nullptr, P3, nullptr, 1.f, 0);
        // scatter gated messages
        hipMemsetAsync(agg, 0, (size_t)kNN * kH * 4, stream);
        k_edge<<<kE, 128, 0, stream>>>(eidx, BIG1, P2, P3, nmlp, agg);
        // h += agg + nmlp
        k_add3<<<(kNN * kH) / 256, 256, 0, stream>>>(h, agg, nmlp);
        // attention
        k_lnrow<<<kNN, 128, 0, stream>>>(h, at_g + mv, at_bt + mv, hnb);
        k_gemm128<<<kNN / 64, 256, 0, stream>>>(hnb, qw + mm, qb + mv, nullptr,
                                                nullptr, nullptr, nullptr, Qb, qscale, 0);
        k_gemm128<<<kNN / 64, 256, 0, stream>>>(hnb, kw + mm, kb + mv, nullptr,
                                                nullptr, nullptr, nullptr, Kb, 1.f, 0);
        k_gemm128<<<kNN / 64, 256, 0, stream>>>(hnb, vw + mm, vb + mv, nullptr,
                                                nullptr, nullptr, nullptr, Vb, 1.f, 0);
        k_scores<<<kB * kNH * 8 * 8, 256, 0, stream>>>(Qb, Kb, BIG1);
        k_softmax<<<kB * kNH * kN, 256, 0, stream>>>(BIG1, BIG2);
        k_pv<<<kB * kNH * 8, 256, 0, stream>>>(BIG2, Vb, Ob);
        // h = O@ow + ob + h  (residual fused); refresh hb
        k_gemm128<<<kNN / 64, 256, 0, stream>>>(Ob, ow + mm, ob + mv, h,
                                                nullptr, nullptr, h, hb, 1.f, 0);
    }

    k_pool<<<kB, 128, 0, stream>>>(h, xmean, xmax);
    k_readout<<<1, 256, 0, stream>>>(xmean, xmax, af,
                                     am_w1, am_b1, am_g, am_bt, am_w2, am_b2,
                                     o_w1, o_b1, o_g1, o_bt1, o_w2, o_b2, o_g2, o_bt2,
                                     o_w3, o_b3, o_w4, o_b4, (float*)d_out);
}
